// ReduceContextAttentionP1_17669495456477
// MI455X (gfx1250) — matrix operand order repaired and checked by reading, not compiled, not run on hardware
//
#include <hip/hip_runtime.h>

// ---------------- problem constants ----------------
#define BATCH  4
#define CDIM   192
#define HDIM   96
#define GDIM   48            // 48x48 patch grid
#define LQ     2304          // GDIM*GDIM  (L == Q)
#define KD     3072          // CDIM * 4 * 4

// ---------------- GEMM tiling ----------------
#define TILE_M 128
#define TILE_N 128
#define BK     32
#define LDSS   40            // padded LDS row stride (halves) to spread banks

typedef __attribute__((ext_vector_type(16))) _Float16 v16h;
typedef __attribute__((ext_vector_type(8)))  _Float16 v8h;
typedef __attribute__((ext_vector_type(8)))  float    v8f;

// -------------------------------------------------------------------------
// 1) per-(b,c) inverse L2 norm of b over HxW
// -------------------------------------------------------------------------
__global__ __launch_bounds__(256) void norm_kernel(const float* __restrict__ bsrc,
                                                   float* __restrict__ invn) {
    __shared__ float red[256];
    const int bc = blockIdx.x;                    // 0 .. BATCH*CDIM-1
    const float* p = bsrc + (size_t)bc * (HDIM * HDIM);
    float s = 0.f;
    for (int i = threadIdx.x; i < HDIM * HDIM; i += 256) {
        float v = p[i];
        s += v * v;
    }
    red[threadIdx.x] = s;
    __syncthreads();
    for (int off = 128; off > 0; off >>= 1) {
        if (threadIdx.x < off) red[threadIdx.x] += red[threadIdx.x + off];
        __syncthreads();
    }
    if (threadIdx.x == 0) invn[bc] = rsqrtf(red[0] + 1e-8f);
}

// -------------------------------------------------------------------------
// 2) patch-mean of (1 - mask): shared by mmk (kernel side) and mmp (query side)
// -------------------------------------------------------------------------
__global__ __launch_bounds__(256) void mmpatch_kernel(const float* __restrict__ mask,
                                                      float* __restrict__ mp) {
    int idx = blockIdx.x * 256 + threadIdx.x;     // b*LQ + l
    if (idx >= BATCH * LQ) return;
    int b = idx / LQ, l = idx - b * LQ;
    int li = l / GDIM, lj = l - li * GDIM;
    const float* mb = mask + (size_t)b * HDIM * HDIM;
    float s = 0.f;
#pragma unroll
    for (int i = 0; i < 4; ++i) {
        int y = li * 2 + i - 1; y = min(max(y, 0), HDIM - 1);
#pragma unroll
        for (int j = 0; j < 4; ++j) {
            int x = lj * 2 + j - 1; x = min(max(x, 0), HDIM - 1);
            s += 1.0f - mb[y * HDIM + x];
        }
    }
    mp[idx] = s * (1.0f / 16.0f);
}

// -------------------------------------------------------------------------
// 3) materialize unfolded patch matrices in f16, row-major over K
//    dst[b][l][k], k = c*16 + i*4 + j ; replicate-pad(1), stride 2, patch 4
// -------------------------------------------------------------------------
__global__ __launch_bounds__(256) void build_patches_kernel(const float* __restrict__ src,
                                                            const float* __restrict__ invn,
                                                            _Float16* __restrict__ dst,
                                                            int use_norm) {
    size_t idx = (size_t)blockIdx.x * 256 + threadIdx.x;   // exact multiple, no guard
    int k = (int)(idx % KD);
    size_t t = idx / KD;
    int l = (int)(t % LQ);
    int b = (int)(t / LQ);
    int c = k >> 4;
    int i = (k >> 2) & 3;
    int j = k & 3;
    int li = l / GDIM, lj = l - li * GDIM;
    int y = li * 2 + i - 1; y = min(max(y, 0), HDIM - 1);
    int x = lj * 2 + j - 1; x = min(max(x, 0), HDIM - 1);
    float v = src[(((size_t)b * CDIM + c) * HDIM + y) * HDIM + x];
    if (use_norm) v *= invn[b * CDIM + c];
    dst[idx] = (_Float16)v;
}

// -------------------------------------------------------------------------
// 4) batched GEMM with WMMA: C[b][m][n] = sum_k A[b][m][k] * B[b][n][k]
//    block tile 128x128, 8 waves in a 4x2 grid, each wave 32x64 output
//    = 2x4 v_wmma_f32_16x16x32_f16 per K-step of 32
// -------------------------------------------------------------------------
__global__ __launch_bounds__(256) void gemm_wmma_kernel(const _Float16* __restrict__ A,
                                                        const _Float16* __restrict__ Bm,
                                                        float* __restrict__ C) {
    __shared__ _Float16 As[TILE_M * LDSS];
    __shared__ _Float16 Bs[TILE_N * LDSS];

    const int batch = blockIdx.z;
    const int m0 = blockIdx.y * TILE_M;
    const int n0 = blockIdx.x * TILE_N;
    const int tid  = threadIdx.x;
    const int lane = tid & 31;
    const int wave = tid >> 5;        // 0..7
    const int wm   = wave >> 1;       // 0..3  (32-row slabs)
    const int wn   = wave & 1;        // 0..1  (64-col slabs)
    const int half = lane >> 4;       // 0/1
    const int r16  = lane & 15;

    const _Float16* Ab = A  + (size_t)batch * LQ * KD;
    const _Float16* Bb = Bm + (size_t)batch * LQ * KD;

    v8f acc[2][4];
#pragma unroll
    for (int mt = 0; mt < 2; ++mt)
#pragma unroll
        for (int nt = 0; nt < 4; ++nt)
            acc[mt][nt] = (v8f){0.f, 0.f, 0.f, 0.f, 0.f, 0.f, 0.f, 0.f};

    // per-thread staging coordinates: 512 16B chunks per 128x32 tile, 2/thread
    const int srow0 = tid >> 2;             // 0..63
    const int sc8   = tid & 3;              // 0..3

    for (int k0 = 0; k0 < KD; k0 += BK) {
        // stage A tile: 128 rows x 32 halves
#pragma unroll
        for (int it = 0; it < 2; ++it) {
            int row = srow0 + it * 64;
            v8h v = *(const v8h*)(Ab + (size_t)(m0 + row) * KD + k0 + sc8 * 8);
            *(v8h*)(As + row * LDSS + sc8 * 8) = v;
        }
        // stage B tile: 128 rows x 32 halves
#pragma unroll
        for (int it = 0; it < 2; ++it) {
            int row = srow0 + it * 64;
            v8h v = *(const v8h*)(Bb + (size_t)(n0 + row) * KD + k0 + sc8 * 8);
            *(v8h*)(Bs + row * LDSS + sc8 * 8) = v;
        }
        // prefetch next K-slab into cache while this tile computes
        if (k0 + BK < KD) {
            __builtin_prefetch(Ab + (size_t)(m0 + srow0) * KD + (k0 + BK) + sc8 * 8, 0, 1);
            __builtin_prefetch(Bb + (size_t)(n0 + srow0) * KD + (k0 + BK) + sc8 * 8, 0, 1);
        }
        __syncthreads();

        // A fragments (16-bit 16x32 layout): lane-half h reads K = {h*8+0..7, 16+h*8+0..7}
        v16h afrag[2];
#pragma unroll
        for (int mt = 0; mt < 2; ++mt) {
            const _Float16* ap = &As[(wm * 32 + mt * 16 + r16) * LDSS + half * 8];
            v8h lo = *(const v8h*)ap;
            v8h hi = *(const v8h*)(ap + 16);
#pragma unroll
            for (int e = 0; e < 8; ++e) { afrag[mt][e] = lo[e]; afrag[mt][e + 8] = hi[e]; }
        }
        // B fragments (32x16 layout): lane-half h reads K = {h*8+0..7, 16+h*8+0..7}
        v16h bfrag[4];
#pragma unroll
        for (int nt = 0; nt < 4; ++nt) {
            const _Float16* bp = &Bs[(wn * 64 + nt * 16 + r16) * LDSS + half * 8];
            v8h lo = *(const v8h*)bp;
            v8h hi = *(const v8h*)(bp + 16);
#pragma unroll
            for (int e = 0; e < 8; ++e) { bfrag[nt][e] = lo[e]; bfrag[nt][e + 8] = hi[e]; }
        }

#pragma unroll
        for (int mt = 0; mt < 2; ++mt)
#pragma unroll
            for (int nt = 0; nt < 4; ++nt)
                acc[mt][nt] = __builtin_amdgcn_wmma_f32_16x16x32_f16(
                    false, afrag[mt], false, bfrag[nt],
                    (short)0, acc[mt][nt], false, false);
        __syncthreads();
    }

    // write back: D layout — VGPR e: row = e + 8*half, col = r16
    float* Cb = C + (size_t)batch * LQ * LQ;
#pragma unroll
    for (int mt = 0; mt < 2; ++mt) {
#pragma unroll
        for (int nt = 0; nt < 4; ++nt) {
            int col = n0 + wn * 64 + nt * 16 + r16;
#pragma unroll
            for (int e = 0; e < 8; ++e) {
                int row = m0 + wm * 32 + mt * 16 + e + half * 8;
                Cb[(size_t)row * LQ + col] = acc[mt][nt][e];
            }
        }
    }
}

// -------------------------------------------------------------------------
// 5) fuse pass 1: diagonal 3x3 sum over the (L,Q) plane, zero pad
// -------------------------------------------------------------------------
__global__ __launch_bounds__(256) void fuse1_kernel(const float* __restrict__ x,
                                                    float* __restrict__ y) {
    int b = blockIdx.z;
    int l = blockIdx.y;
    int q = blockIdx.x * 256 + threadIdx.x;
    const float* xb = x + (size_t)b * LQ * LQ;
    float s = xb[(size_t)l * LQ + q];
    if (l > 0 && q > 0)           s += xb[(size_t)(l - 1) * LQ + (q - 1)];
    if (l < LQ - 1 && q < LQ - 1) s += xb[(size_t)(l + 1) * LQ + (q + 1)];
    y[(size_t)b * LQ * LQ + (size_t)l * LQ + q] = s;
}

// -------------------------------------------------------------------------
// 6) fuse pass 2: diagonal sum in the transposed flat indexing, written back
//    in original layout (no explicit transpose buffers needed)
// -------------------------------------------------------------------------
__global__ __launch_bounds__(256) void fuse2_kernel(const float* __restrict__ x,
                                                    float* __restrict__ y) {
    int b = blockIdx.z;
    int l = blockIdx.y;
    int q = blockIdx.x * 256 + threadIdx.x;
    int i = l / GDIM, j = l - i * GDIM;       // l = i*GDIM + j
    int u = q / GDIM, v = q - u * GDIM;       // q = u*GDIM + v
    int lt = j * GDIM + i;                    // transposed flat index
    int qt = v * GDIM + u;
    const float* xb = x + (size_t)b * LQ * LQ;
    float s = 0.f;
#pragma unroll
    for (int d = -1; d <= 1; ++d) {
        int a  = lt + d;
        int bb = qt + d;
        if (a >= 0 && a < LQ && bb >= 0 && bb < LQ) {
            int j2 = a / GDIM, i2 = a - j2 * GDIM;     // a = j2*GDIM + i2
            int v2 = bb / GDIM, u2 = bb - v2 * GDIM;   // bb = v2*GDIM + u2
            s += xb[(size_t)(i2 * GDIM + j2) * LQ + (u2 * GDIM + v2)];
        }
    }
    y[(size_t)b * LQ * LQ + (size_t)l * LQ + q] = s;
}

// -------------------------------------------------------------------------
// 7) mask + scale + softmax over L per (b,q) column; online max/sum,
//    64 coalesced columns per block, 4-way partial merge in LDS
// -------------------------------------------------------------------------
__global__ __launch_bounds__(256) void softmax_kernel(const float* __restrict__ x,
                                                      const float* __restrict__ mmpatch,
                                                      float* __restrict__ out) {
    __shared__ float sm[256];
    __shared__ float ss[256];
    const int b  = blockIdx.y;
    const int qi = threadIdx.x & 63;
    const int ls = threadIdx.x >> 6;          // 0..3
    const int q  = blockIdx.x * 64 + qi;
    const float* xb  = x + (size_t)b * LQ * LQ;
    const float* mpB = mmpatch + b * LQ;
    const float mmp_q = mpB[q];
    const bool  qcond = mmp_q > 0.5f;

    float m = -1e30f, s = 0.f;
    for (int l = ls; l < LQ; l += 4) {
        float mk = mpB[l];
        float g = (((mk > mmp_q) && qcond) || (mk == 1.0f)) ? 1.f : 0.f;
        float v = xb[(size_t)l * LQ + q] * g * 10.0f;
        if (v > m) { s = s * __expf(m - v) + 1.f; m = v; }
        else       { s += __expf(v - m); }
    }
    sm[threadIdx.x] = m;
    ss[threadIdx.x] = s;
    __syncthreads();
    for (int off = 128; off >= 64; off >>= 1) {
        if (threadIdx.x < off) {
            float m1 = sm[threadIdx.x], s1 = ss[threadIdx.x];
            float m2 = sm[threadIdx.x + off], s2 = ss[threadIdx.x + off];
            float mn = fmaxf(m1, m2);
            ss[threadIdx.x] = s1 * __expf(m1 - mn) + s2 * __expf(m2 - mn);
            sm[threadIdx.x] = mn;
        }
        __syncthreads();
    }
    const float M = sm[qi];
    const float inv = 1.0f / ss[qi];
    float* ob = out + (size_t)b * LQ * LQ;
    for (int l = ls; l < LQ; l += 4) {
        float mk = mpB[l];
        float g = (((mk > mmp_q) && qcond) || (mk == 1.0f)) ? 1.f : 0.f;
        float v = xb[(size_t)l * LQ + q] * g * 10.0f;
        ob[(size_t)l * LQ + q] = __expf(v - M) * inv;
    }
}

// -------------------------------------------------------------------------
// host launcher
// -------------------------------------------------------------------------
extern "C" void kernel_launch(void* const* d_in, const int* in_sizes, int n_in,
                              void* d_out, int out_size, void* d_ws, size_t ws_size,
                              hipStream_t stream) {
    (void)in_sizes; (void)n_in; (void)out_size; (void)ws_size;
    const float* f    = (const float*)d_in[0];
    const float* bsrc = (const float*)d_in[1];
    const float* mask = (const float*)d_in[2];
    float* out = (float*)d_out;

    // workspace carve (256B aligned slabs)
    size_t off = 0;
    char* ws = (char*)d_ws;
    auto carve = [&](size_t bytes) -> void* {
        void* p = ws + off;
        off += (bytes + 255) & ~(size_t)255;
        return p;
    };
    _Float16* A_h  = (_Float16*)carve((size_t)BATCH * LQ * KD * sizeof(_Float16)); // kernel (bn patches)
    _Float16* B_h  = (_Float16*)carve((size_t)BATCH * LQ * KD * sizeof(_Float16)); // query  (f patches)
    float*    cosb = (float*)carve((size_t)BATCH * LQ * LQ * sizeof(float));
    float*    tmpb = (float*)carve((size_t)BATCH * LQ * LQ * sizeof(float));
    float*    invn = (float*)carve((size_t)BATCH * CDIM * sizeof(float));
    float*    mp   = (float*)carve((size_t)BATCH * LQ * sizeof(float));

    // 1) per-(b,c) inverse norm of b
    norm_kernel<<<BATCH * CDIM, 256, 0, stream>>>(bsrc, invn);

    // 2) shared patch-mean of (1 - mask)
    mmpatch_kernel<<<(BATCH * LQ + 255) / 256, 256, 0, stream>>>(mask, mp);

    // 3) build f16 patch matrices
    const int buildBlocks = (int)(((size_t)BATCH * LQ * KD) / 256);   // exact
    build_patches_kernel<<<buildBlocks, 256, 0, stream>>>(bsrc, invn, A_h, 1);
    build_patches_kernel<<<buildBlocks, 256, 0, stream>>>(f,    invn, B_h, 0);

    // 4) batched WMMA GEMM -> cos[b][l][q]
    dim3 ggrid(LQ / TILE_N, LQ / TILE_M, BATCH);   // (18, 18, 4)
    gemm_wmma_kernel<<<ggrid, 256, 0, stream>>>(A_h, B_h, cosb);

    // 5/6) fuse passes: cos -> tmp -> cos
    dim3 fgrid(LQ / 256, LQ, BATCH);               // (9, 2304, 4)
    fuse1_kernel<<<fgrid, 256, 0, stream>>>(cosb, tmpb);
    fuse2_kernel<<<fgrid, 256, 0, stream>>>(tmpb, cosb);

    // 7) mask + scale + softmax over L -> out
    dim3 sgrid(LQ / 64, BATCH);                    // (36, 4)
    softmax_kernel<<<sgrid, 256, 0, stream>>>(cosb, mp, out);
}
